// E3ConvLayer_17806934409755
// MI455X (gfx1250) — matrix operand order repaired and checked by reading, not compiled, not run on hardware
//
#include <hip/hip_runtime.h>
#include <hip/hip_bf16.h>

typedef __attribute__((ext_vector_type(16))) __bf16 v16bf;
typedef __attribute__((ext_vector_type(8)))  float  v8f;
typedef __attribute__((ext_vector_type(4)))  unsigned int v4u;
typedef __attribute__((ext_vector_type(8)))  int v8i;
typedef __attribute__((ext_vector_type(4)))  int v4i;

#define NA      50000
#define MN      32
#define AF      128
#define NF      64
#define ETOT    (NA*MN)        // 1,600,000 edges
#define NTILES  (ETOT/16)      // 100,000 tiles of 16 edges

// workspace layout (bytes). The four packs are contiguous: 80 KB total.
#define OFF_CNT   0                       // NA floats (200,000 B)
#define OFF_TPWH  (256*1024)              // 16384 bf16 = 32 KB
#define OFF_TPWL  (OFF_TPWH + 32768)      // 32 KB
#define OFF_W1H   (OFF_TPWL + 32768)      // 4096 bf16 = 8 KB
#define OFF_W1L   (OFF_W1H + 8192)        // 8 KB
#define PACK_BYTES 81920                  // tpwh+tpwl+w1h+w1l
#define PACK_DW    (PACK_BYTES/4)         // 20480 dwords

// LDS pack layout in v16bf (32 B) units
#define L_TPWH 0
#define L_TPWL 1024
#define L_W1H  2048
#define L_W1L  2304
#define L_TOT  2560                        // 80 KB

// ---------------------------------------------------------------------------
__global__ void e3_zero(float* __restrict__ out, float* __restrict__ cnt) {
    int i = blockIdx.x * blockDim.x + threadIdx.x;
    if (i < NA * AF) out[i] = 0.0f;
    if (i < NA)      cnt[i] = 0.0f;
}

// ---------------------------------------------------------------------------
// Pack tp_w (128x128) and w1 (64x64) into bf16 hi/lo WMMA B-operand layout.
// B-matrix 32x16 (KxN), in this code's contiguous k order: lane L -> column n = L%16; lanes 0-15 hold
// K = kc*32 + 0..15, lanes 16-31 hold K = kc*32 + 16..31; element j -> K+j.
__global__ void e3_prep(const float* __restrict__ w1, const float* __restrict__ tpw,
                        __bf16* __restrict__ tpwh, __bf16* __restrict__ tpwl,
                        __bf16* __restrict__ w1h,  __bf16* __restrict__ w1l) {
    int t = blockIdx.x * blockDim.x + threadIdx.x;
    if (t < 4 * 8 * 32 * 16) {      // tp_w pack: [kc(4)][ct(8)][lane(32)][j(16)]
        int j    = t & 15;
        int lane = (t >> 4) & 31;
        int ct   = (t >> 9) & 7;
        int kc   = t >> 12;
        int hf = lane >> 4, nl = lane & 15;
        int k = kc * 32 + hf * 16 + j;
        int n = ct * 16 + nl;
        float x = tpw[k * AF + n];
        __bf16 h = (__bf16)x;
        tpwh[t] = h;
        tpwl[t] = (__bf16)(x - (float)h);
    }
    if (t < 2 * 4 * 32 * 16) {      // w1 pack: [kc(2)][ct(4)][lane(32)][j(16)]
        int j    = t & 15;
        int lane = (t >> 4) & 31;
        int ct   = (t >> 9) & 3;
        int kc   = t >> 11;
        int hf = lane >> 4, nl = lane & 15;
        int k = kc * 32 + hf * 16 + j;
        int n = ct * 16 + nl;
        float x = w1[k * NF + n];
        __bf16 h = (__bf16)x;
        w1h[t] = h;
        w1l[t] = (__bf16)(x - (float)h);
    }
}

// ---------------------------------------------------------------------------
// Build bf16 hi/lo A-operand for one 32-wide K chunk.
// 16-bit A 16x32 layout: lane L -> row m = L%16;
// element j -> K = kb+j, where the caller's kb includes 16*(L/16) (contiguous order)
__device__ __forceinline__ void build_a(const float* __restrict__ row, int kb,
                                        v16bf& ah, v16bf& al) {
    float4 f0 = *(const float4*)(row + kb);
    float4 f1 = *(const float4*)(row + kb + 4);
    float4 f2 = *(const float4*)(row + kb + 8);
    float4 f3 = *(const float4*)(row + kb + 12);
    float xs[16] = {f0.x, f0.y, f0.z, f0.w, f1.x, f1.y, f1.z, f1.w,
                    f2.x, f2.y, f2.z, f2.w, f3.x, f3.y, f3.z, f3.w};
#pragma unroll
    for (int j = 0; j < 16; ++j) {
        __bf16 h = (__bf16)xs[j];
        ah[j] = h;
        al[j] = (__bf16)(xs[j] - (float)h);
    }
}

// bf16x3 split-precision accumulate: C += Ahi*Bhi + Ahi*Blo + Alo*Bhi
__device__ __forceinline__ v8f wmma3(v16bf ah, v16bf al, v16bf bh, v16bf bl, v8f c) {
    c = __builtin_amdgcn_wmma_f32_16x16x32_bf16(false, ah, false, bh, (short)0, c, false, false);
    c = __builtin_amdgcn_wmma_f32_16x16x32_bf16(false, ah, false, bl, (short)0, c, false, false);
    c = __builtin_amdgcn_wmma_f32_16x16x32_bf16(false, al, false, bh, (short)0, c, false, false);
    return c;
}

// ---------------------------------------------------------------------------
// Main fused kernel: 128 threads = 4 waves; each wave handles 4 tiles of 16
// edges. The 80 KB weight pack is staged into LDS once per WG via the Tensor
// Data Mover (falls back to a cooperative vector copy if the builtin is absent).
__global__ __launch_bounds__(128, 1)
void e3_main(const float* __restrict__ atom_fea, const float* __restrict__ nbr_fea,
             const int* __restrict__ nbr_idx,
             const float* __restrict__ w2, const float* __restrict__ b1,
             const float* __restrict__ b2,
             const __bf16* __restrict__ pack_g,   // contiguous 80 KB pack in ws
             float* __restrict__ out_sums, float* __restrict__ cnt) {
    __shared__ v16bf lds_pack[L_TOT];   // 80 KB

    const int lane = threadIdx.x & 31;
    const int wave = threadIdx.x >> 5;

#if defined(__gfx1250__) && __has_builtin(__builtin_amdgcn_tensor_load_to_lds) && \
    __has_builtin(__builtin_amdgcn_s_wait_tensorcnt)
    if (wave == 0) {
        // D# per CDNA5 ISA: one 2D tile, 1 row of
        // 20480 x 4-byte elements (80 KB), global -> LDS.
        unsigned long long ga = (unsigned long long)(size_t)pack_g;
        unsigned int ldsAddr = (unsigned int)(size_t)&lds_pack[0];  // low 32 bits = LDS offset
        v4u g0;
        g0[0] = 1u;                                     // count=1, user descriptor
        g0[1] = ldsAddr;                                // lds_addr [63:32]
        g0[2] = (unsigned int)(ga & 0xFFFFFFFFull);     // global_addr [95:64]
        g0[3] = (unsigned int)((ga >> 32) & 0x1FFFFFFull) | (2u << 30);  // addr hi + type=2
        v8i g1;
        g1[0] = 0x00020000;                 // workgroup_mask=0, data_size=2 (4 B)
        g1[1] = (int)(PACK_DW << 16);       // tensor_dim0[15:0] at bits[63:48]
        g1[2] = (int)(1u << 16);            // tensor_dim1 = 1 at bits[111:80]
        g1[3] = (int)(PACK_DW << 16);       // tile_dim0 at bits[127:112]
        g1[4] = 1;                          // tile_dim1 = 1
        g1[5] = (int)PACK_DW;               // tensor_dim0_stride[31:0]
        g1[6] = 0;
        g1[7] = 0;
        v4i gz4 = {0, 0, 0, 0};
        v8i gz8 = {0, 0, 0, 0, 0, 0, 0, 0};
        __builtin_amdgcn_tensor_load_to_lds(g0, g1, gz4, gz4, gz8, 0);
        __builtin_amdgcn_s_wait_tensorcnt((short)0);    // TENSORcnt == 0
    }
#else
    {   // cooperative fallback copy: 5120 uint4s
        uint4* dst = (uint4*)(&lds_pack[0]);
        const uint4* src = (const uint4*)pack_g;
        for (int i = threadIdx.x; i < PACK_BYTES / 16; i += blockDim.x) dst[i] = src[i];
    }
#endif
    __syncthreads();

    const int hf = lane >> 4;       // half of the wave
    const int nl = lane & 15;       // lane within half == tile row / column

    for (int t = 0; t < 4; ++t) {
        const int tile = (blockIdx.x * 4 + wave) * 4 + t;
        if (tile >= NTILES) continue;
        const int eBase = tile * 16;
        const int eOwn  = eBase + nl;             // this lane's edge row
        const int idxOwn = nbr_idx[eOwn];         // gathered atom index

        // ---------------- Stage A: radial MLP scalar ----------------
        const float* rrow = nbr_fea + (size_t)eOwn * NF;
        v8f accA[4] = {};
#pragma unroll
        for (int kc = 0; kc < 2; ++kc) {
            v16bf ah, al;
            build_a(rrow, kc * 32 + hf * 16, ah, al);
#pragma unroll
            for (int ct = 0; ct < 4; ++ct) {
                v16bf bh = lds_pack[L_W1H + (kc * 4 + ct) * 32 + lane];
                v16bf bl = lds_pack[L_W1L + (kc * 4 + ct) * 32 + lane];
                accA[ct] = wmma3(ah, al, bh, bl, accA[ct]);
            }
        }
        // softplus(h + b1) dotted with w2[:,0] in the C-tile layout:
        // lane holds column n = ct*16+nl for rows m = r + 8*hf
        float w2c[4], b1v[4];
#pragma unroll
        for (int ct = 0; ct < 4; ++ct) {
            int n = ct * 16 + nl;
            w2c[ct] = w2[n * 9];      // SH_DIM = 9, column 0
            b1v[ct] = b1[n];
        }
        float part[8];
#pragma unroll
        for (int r = 0; r < 8; ++r) {
            float s = 0.0f;
#pragma unroll
            for (int ct = 0; ct < 4; ++ct) {
                float h = accA[ct][r] + b1v[ct];
                float sp = (h > 20.0f) ? h : log1pf(__expf(h));
                s += sp * w2c[ct];
            }
            part[r] = s;
        }
        // butterfly-reduce across the 16 lanes of each half: afterwards every
        // lane holds the full scalars for its half's 8 rows (m = r + 8*hf)
#pragma unroll
        for (int m = 1; m < 16; m <<= 1) {
#pragma unroll
            for (int r = 0; r < 8; ++r) part[r] += __shfl_xor(part[r], m, 16);
        }
        const float b20 = b2[0];
        const float invs = 0.0883883476483184f;   // 1/sqrt(128)
        float scal[8];
#pragma unroll
        for (int r = 0; r < 8; ++r) scal[r] = (part[r] + b20) * invs;

        // destination atom rows for the 8 output rows this lane scatters
        int rowIdx[8];
#pragma unroll
        for (int r = 0; r < 8; ++r) rowIdx[r] = __shfl(idxOwn, hf * 8 + r, 32);

        // ---------------- Stage B: neigh @ tp_w ----------------
        const float* arow = atom_fea + (size_t)idxOwn * AF;
        v8f acc[8] = {};
#pragma unroll
        for (int kc = 0; kc < 4; ++kc) {
            v16bf ah, al;
            build_a(arow, kc * 32 + hf * 16, ah, al);
#pragma unroll
            for (int ct = 0; ct < 8; ++ct) {
                v16bf bh = lds_pack[L_TPWH + (kc * 8 + ct) * 32 + lane];
                v16bf bl = lds_pack[L_TPWL + (kc * 8 + ct) * 32 + lane];
                acc[ct] = wmma3(ah, al, bh, bl, acc[ct]);
            }
        }

        // scale by per-row scalar and atomically scatter (segment sum)
#pragma unroll
        for (int ct = 0; ct < 8; ++ct) {
            const int col = ct * 16 + nl;
#pragma unroll
            for (int r = 0; r < 8; ++r) {
                float v = acc[ct][r] * scal[r];
                unsafeAtomicAdd(&out_sums[(size_t)rowIdx[r] * AF + col], v);
            }
        }
        // neighbor counts: one add per edge (lanes 0-15 own the 16 edges)
        if (hf == 0) unsafeAtomicAdd(&cnt[idxOwn], 1.0f);
    }
}

// ---------------------------------------------------------------------------
__global__ void e3_finalize(float* __restrict__ out, const float* __restrict__ cnt) {
    int i = blockIdx.x * blockDim.x + threadIdx.x;
    if (i < NA * AF) {
        float c = cnt[i >> 7];
        out[i] = out[i] / fmaxf(c, 1.0f);
    }
}

// ---------------------------------------------------------------------------
extern "C" void kernel_launch(void* const* d_in, const int* in_sizes, int n_in,
                              void* d_out, int out_size, void* d_ws, size_t ws_size,
                              hipStream_t stream) {
    const float* atom_fea = (const float*)d_in[0];
    const float* nbr_fea  = (const float*)d_in[1];
    const int*   nbr_idx  = (const int*)d_in[2];
    const float* w1       = (const float*)d_in[3];
    const float* b1       = (const float*)d_in[4];
    const float* w2       = (const float*)d_in[5];
    const float* b2       = (const float*)d_in[6];
    const float* tpw      = (const float*)d_in[7];
    float* out = (float*)d_out;
    char*  ws  = (char*)d_ws;

    float*  cnt  = (float*)(ws + OFF_CNT);
    __bf16* tpwh = (__bf16*)(ws + OFF_TPWH);
    __bf16* tpwl = (__bf16*)(ws + OFF_TPWL);
    __bf16* w1h  = (__bf16*)(ws + OFF_W1H);
    __bf16* w1l  = (__bf16*)(ws + OFF_W1L);

    e3_zero<<<(NA * AF + 255) / 256, 256, 0, stream>>>(out, cnt);
    e3_prep<<<(16384 + 255) / 256, 256, 0, stream>>>(w1, tpw, tpwh, tpwl, w1h, w1l);

    const int nWG = (NTILES + 15) / 16;    // 4 waves * 4 tiles per WG
    e3_main<<<nWG, 128, 0, stream>>>(atom_fea, nbr_fea, nbr_idx, w2, b1, b2,
                                     tpwh, out, cnt);

    e3_finalize<<<(NA * AF + 255) / 256, 256, 0, stream>>>(out, cnt);
}
